// EncoderByType_7121055776914
// MI455X (gfx1250) — compile-verified, matrix operand order checked by reading, not run on hardware
//
#include <hip/hip_runtime.h>
#include <hip/hip_bf16.h>

typedef __attribute__((ext_vector_type(16))) __bf16 v16bf;
typedef __attribute__((ext_vector_type(8)))  __bf16 v8bf;
typedef __attribute__((ext_vector_type(4)))  __bf16 v4bf;
typedef __attribute__((ext_vector_type(8)))  float  v8f;
typedef __attribute__((ext_vector_type(4)))  float  v4f;

// ---------------------------------------------------------------------------
// K-offset mapping for the 16-bit 16x32 A / 32x16 B WMMA fragment layout:
//   lanes 0-15 : K in {0..7} U {16..23},  lanes 16-31 : K in {8..15} U {24..31}
//   element e (0..15): vgpr v=e/2, pair p=e%2 ->
//   k = ((v<4)?0:16) + half*8 + (v&3)*2 + p
// ---------------------------------------------------------------------------
__device__ __host__ __forceinline__ int frag_k(int half, int e) {
    int v = e >> 1, p = e & 1;
    return ((v < 4) ? 0 : 16) + half * 8 + ((v & 3) << 1) + p;
}

// Pack f32 weight W[K x N] (row-major, fan_in x fan_out) into bf16 fragments:
// dst layout: [kt][nt][lane 0..31][e 0..15]
__global__ void pack_weights_kernel(const float* __restrict__ W,
                                    __bf16* __restrict__ dst, int K, int N) {
    int idx = blockIdx.x * blockDim.x + threadIdx.x;
    int total = K * N;
    if (idx >= total) return;
    int e    = idx & 15;
    int lane = (idx >> 4) & 31;
    int t    = idx >> 9;            // kt * numNT + nt
    int numNT = N >> 4;
    int nt = t % numNT;
    int kt = t / numNT;
    int half = lane >> 4;
    int n = nt * 16 + (lane & 15);
    int k = kt * 32 + frag_k(half, e);
    dst[idx] = (__bf16)W[(long long)k * N + n];
}

// ---------------------------------------------------------------------------
// Fragment loaders
// ---------------------------------------------------------------------------
// A fragment from per-wave LDS activation buffer (16 rows x K, row-major bf16)
__device__ __forceinline__ v16bf a_from_lds(const __bf16* base, int K,
                                            int lane, int kt) {
    int row  = lane & 15;
    int half = lane >> 4;
    const __bf16* p = base + row * K + kt * 32 + half * 8;
    union { v16bf v; v8bf h[2]; } u;
    u.h[0] = *(const v8bf*)p;            // ds_load_b128
    u.h[1] = *(const v8bf*)(p + 16);     // ds_load_b128
    return u.v;
}

// B fragment from pre-packed bf16 weights (L2/WGP$-resident, shared by waves)
__device__ __forceinline__ v16bf b_frag(const __bf16* __restrict__ wf,
                                        int numNT, int kt, int nt, int lane) {
    const __bf16* p = wf + ((((long long)kt * numNT + nt) * 32 + lane) << 4);
    union { v16bf v; v8bf h[2]; } u;
    u.h[0] = *(const v8bf*)p;            // global_load_b128
    u.h[1] = *(const v8bf*)(p + 8);      // global_load_b128
    return u.v;
}

__device__ __forceinline__ float sigmoidf_fast(float v) {
    return 1.0f / (1.0f + __expf(-v));
}

// Stage this wave's 16 contiguous rows of x (f32) into LDS as bf16, once.
// The 16 x dIn tile is a flat contiguous range of x -> fully coalesced b128.
__device__ __forceinline__ void stage_x(const float* __restrict__ x,
                                        long long rowBase, int dIn,
                                        __bf16* dst, int lane) {
    const float* src = x + rowBase * (long long)dIn;
    const int total4 = (16 * dIn) >> 2;
    for (int t = lane; t < total4; t += 32) {
        v4f f = *(const v4f*)(src + (t << 2));
        v4bf h;
#pragma unroll
        for (int i = 0; i < 4; ++i) h[i] = (__bf16)f[i];
        *(v4bf*)(dst + (t << 2)) = h;    // ds_store_b64
    }
}

// ---------------------------------------------------------------------------
// One Linear+Sigmoid layer for this wave's 16 rows (A always from LDS).
//   OG: result goes to global f32 out, else to LDS bf16 buffer.
// N processed in groups of 8 column-tiles: each A fragment feeds 8 WMMAs.
// ---------------------------------------------------------------------------
template <bool OG>
__device__ __forceinline__ void mlp_layer(const __bf16* aBuf, int K, int N,
                                          const __bf16* __restrict__ wf,
                                          const float* __restrict__ bias,
                                          __bf16* oBuf, float* __restrict__ og,
                                          int oStride, long long rowBase,
                                          int lane) {
    const int numKT = K >> 5;
    const int numNT = N >> 4;
    const int col0  = lane & 15;
    const int half  = lane >> 4;
    const v8f zero = {0.f, 0.f, 0.f, 0.f, 0.f, 0.f, 0.f, 0.f};

    for (int ng = 0; ng < numNT; ng += 8) {
        v8f acc[8];
#pragma unroll
        for (int i = 0; i < 8; ++i) acc[i] = zero;

        for (int kt = 0; kt < numKT; ++kt) {
            v16bf a = a_from_lds(aBuf, K, lane, kt);
#pragma unroll
            for (int i = 0; i < 8; ++i) {
                v16bf b = b_frag(wf, numNT, kt, ng + i, lane);
                acc[i] = __builtin_amdgcn_wmma_f32_16x16x32_bf16(
                    /*neg_a=*/false, a, /*neg_b=*/false, b,
                    /*c_mod=*/(short)0, acc[i],
                    /*reuse_a=*/false, /*reuse_b=*/false);
            }
        }

#pragma unroll
        for (int i = 0; i < 8; ++i) {
            int col = (ng + i) * 16 + col0;
            float bs = bias[col];
#pragma unroll
            for (int r = 0; r < 8; ++r) {
                float vv = sigmoidf_fast(acc[i][r] + bs);
                int m = r + 8 * half;   // row within the 16-row tile
                if (OG) {
                    og[(rowBase + m) * (long long)oStride + col] = vv;
                } else {
                    oBuf[m * N + col] = (__bf16)vv;
                }
            }
        }
    }
}

// ---------------------------------------------------------------------------
// Fused 3-layer MLP: 8 waves/block, each wave owns 16 rows end-to-end.
// Per-wave LDS: two 16x512 bf16 ping-pong buffers (32 KB) -> 256 KB/block.
// x tile is staged once into bufB (idle during layer 1), then:
//   L1: bufB(x, K=dIn) -> bufA    L2: bufA -> bufB    L3: bufB -> global
// No cross-wave barriers: each wave's LDS slice is private.
// ---------------------------------------------------------------------------
__global__ void __launch_bounds__(256, 1)
mlp_fused_kernel(const float* __restrict__ x, int dIn, long long nRows,
                 const __bf16* __restrict__ w1f, const float* __restrict__ b1,
                 const __bf16* __restrict__ w2f, const float* __restrict__ b2,
                 const __bf16* __restrict__ w3f, const float* __restrict__ b3,
                 float* __restrict__ out) {
    extern __shared__ __bf16 smem[];
    const int lane = threadIdx.x & 31;
    const int wid  = threadIdx.x >> 5;
    long long rowBase = (long long)blockIdx.x * 128 + wid * 16;
    if (rowBase >= nRows) return;   // wave-uniform; row counts are 16-aligned

    __bf16* bufA = smem + (size_t)wid * 16384;  // 16 x 512
    __bf16* bufB = bufA + 8192;

    // Stage x tile (16 x dIn, contiguous in global) as bf16 into bufB.
    stage_x(x, rowBase, dIn, bufB, lane);

    // L1: bufB (K=dIn) -> bufA (16x512 bf16)
    mlp_layer<false>(bufB, dIn, 512, w1f, b1, bufA, nullptr, 0, rowBase, lane);
    // L2: bufA (K=512) -> bufB (16x512 bf16)
    mlp_layer<false>(bufA, 512, 512, w2f, b2, bufB, nullptr, 0, rowBase, lane);
    // L3: bufB (K=512) -> global out (N=256, f32)
    mlp_layer<true>(bufB, 512, 256, w3f, b3, nullptr, out, 256, rowBase, lane);
}

// ---------------------------------------------------------------------------
extern "C" void kernel_launch(void* const* d_in, const int* in_sizes, int n_in,
                              void* d_out, int out_size, void* d_ws, size_t ws_size,
                              hipStream_t stream) {
    const int dims[3] = {64, 128, 256};
    const float* xs[3];
    long long counts[3];
    for (int t = 0; t < 3; ++t) {
        xs[t] = (const float*)d_in[t];
        counts[t] = (long long)in_sizes[t] / dims[t];
    }
    // inputs: 0..2 = x0..x2, 3 = node_type, then W{t}_{1..3}/b{t}_{1..3}
    const float* W[3][3];
    const float* B[3][3];
    for (int t = 0; t < 3; ++t)
        for (int l = 0; l < 3; ++l) {
            W[t][l] = (const float*)d_in[4 + t * 6 + l * 2];
            B[t][l] = (const float*)d_in[4 + t * 6 + l * 2 + 1];
        }

    // Pack all 9 weight matrices into bf16 fragment layout in d_ws (~2.8 MB).
    __bf16* wf[3][3];
    size_t off = 0;
    for (int t = 0; t < 3; ++t) {
        int Ks[3] = {dims[t], 512, 512};
        int Ns[3] = {512, 512, 256};
        for (int l = 0; l < 3; ++l) {
            wf[t][l] = (__bf16*)d_ws + off;
            size_t total = (size_t)Ks[l] * Ns[l];
            off += total;
            int blocks = (int)((total + 255) / 256);
            pack_weights_kernel<<<blocks, 256, 0, stream>>>(
                W[t][l], wf[t][l], Ks[l], Ns[l]);
        }
    }

    // node_type is sorted, so output rows are simply concatenated per type.
    long long rowOff = 0;
    for (int t = 0; t < 3; ++t) {
        int blocks = (int)((counts[t] + 127) / 128);
        mlp_fused_kernel<<<blocks, 256, 262144, stream>>>(
            xs[t], dims[t], counts[t],
            wf[t][0], B[t][0], wf[t][1], B[t][1], wf[t][2], B[t][2],
            (float*)d_out + rowOff * 256);
        rowOff += counts[t];
    }
}
